// AutoregressiveLSTM_84198538871345
// MI455X (gfx1250) — compile-verified, matrix operand order checked by reading, not run on hardware
//
#include <hip/hip_runtime.h>
#include <hip/hip_bf16.h>
#include <stdint.h>

// ---------------------------------------------------------------------------
// AutoregressiveLSTM on MI455X (gfx1250): bf16 WMMA 16x16x32 recurrent GEMMs,
// LDS-staged weight slabs (XOR-swizzled, double-buffered), fp32 cell state,
// fused gate epilogue, ping-pong h buffers, stream-sequenced steps.
// ---------------------------------------------------------------------------

typedef __bf16 bf16_t;
typedef __attribute__((ext_vector_type(8)))  __bf16 v8bf;
typedef __attribute__((ext_vector_type(16))) __bf16 v16bf;
typedef __attribute__((ext_vector_type(8)))  float  v8f;

#define B_    512
#define T_    256
#define I_    256
#define H_    1024
#define V_    1024
#define E_    8
#define FUT_  64
#define KENC  1280   // I_ + H_
#define KDEC  1056   // H_ + 32 (E_=8 zero-padded to one 32-wide WMMA slab)

// ---------------------------------------------------------------------------
// Global fragment loader for 16-bit WMMA operands from row-major [rows, ld].
// A-matrix 16x32 bf16 layout (ISA 7.12.2): lane L holds row (L&15); lanes 0-15
// hold K = {k0..k0+7, k0+16..k0+23}, lanes 16-31 hold K = {k0+8..k0+15,
// k0+24..k0+31}.  Two contiguous 16-byte chunks -> two global_load_b128.
// ---------------------------------------------------------------------------
__device__ __forceinline__ v16bf load_frag(const bf16_t* __restrict__ p,
                                           int row0, int ld, int k0) {
  const int lane = threadIdx.x & 31;
  const bf16_t* q = p + (size_t)(row0 + (lane & 15)) * (size_t)ld
                      + (size_t)(k0 + ((lane >> 4) << 3));
  v8bf lo = *(const v8bf*)(q);
  v8bf hi = *(const v8bf*)(q + 16);
  return __builtin_shufflevector(lo, hi,
           0, 1, 2, 3, 4, 5, 6, 7, 8, 9, 10, 11, 12, 13, 14, 15);
}

// ---------------------------------------------------------------------------
// LDS slab: ROWS x 32 bf16, stored as 4 swizzled 16B chunks per row.
// Chunk c of row r lives at byte offset for chunk (c ^ ((r>>2)&3)); the four
// rows that share a bank group (r, r+4, r+8, r+12) then hit disjoint chunks,
// making the two ds_load_b128 per fragment bank-conflict-free at stride 64B.
// ---------------------------------------------------------------------------
__device__ __forceinline__ int lds_off(int row, int chunk) {
  return (row << 5) + (((chunk ^ ((row >> 2) & 3)) << 3));
}

// Fetch one slab's worth of weight chunks into registers.  ROWS/64 chunks per
// thread; row r of the slab maps to weight row (row_base + (r>>7)*H_ + (r&127))
// so the 4-gate cell slabs (ROWS=512) and the plain fc slab (ROWS=128) share
// this helper.
template <int ROWS>
__device__ __forceinline__ void fetch_slab(v8bf* st,
                                           const bf16_t* __restrict__ W,
                                           int ld, int row_base, int k) {
  const int tid = threadIdx.x;
#pragma unroll
  for (int j = 0; j < ROWS / 64; ++j) {
    const int cid = tid + j * 256;
    const int r = cid >> 2;
    const int c = cid & 3;
    const int srow = row_base + ((r >> 7) * H_) + (r & 127);
    st[j] = *(const v8bf*)(W + (size_t)srow * (size_t)ld + k + (c << 3));
  }
}

template <int ROWS>
__device__ __forceinline__ void store_slab(bf16_t* sm, const v8bf* st) {
  const int tid = threadIdx.x;
#pragma unroll
  for (int j = 0; j < ROWS / 64; ++j) {
    const int cid = tid + j * 256;
    const int r = cid >> 2;
    const int c = cid & 3;
    *(v8bf*)(sm + lds_off(r, c)) = st[j];
  }
}

// Fragment from a staged slab (row0 is the row inside the slab).
__device__ __forceinline__ v16bf frag_from_lds(const bf16_t* sm, int row0) {
  const int lane = threadIdx.x & 31;
  const int row = row0 + (lane & 15);
  const int c0 = lane >> 4;  // 0 or 1
  v8bf lo = *(const v8bf*)(sm + lds_off(row, c0));
  v8bf hi = *(const v8bf*)(sm + lds_off(row, c0 + 2));
  return __builtin_shufflevector(lo, hi,
           0, 1, 2, 3, 4, 5, 6, 7, 8, 9, 10, 11, 12, 13, 14, 15);
}

__device__ __forceinline__ v8f wmma_bf16(v16bf a, v16bf b, v8f c) {
  return __builtin_amdgcn_wmma_f32_16x16x32_bf16(
      /*neg_a=*/false, a, /*neg_b=*/false, b,
      /*c_mod=*/(short)0, c, /*reuse_a=*/false, /*reuse_b=*/false);
}

__device__ __forceinline__ float sigmoidf_(float x) {
  return 1.0f / (1.0f + __expf(-x));
}

// ---------------------------------------------------------------------------
// One encoder LSTM step.  z = [x_t, h] @ Wenc^T + b, gates, update (c, h).
// Grid (8, 8): blockIdx.x -> 128 cols of H, blockIdx.y -> 64 rows of B.
// 8 waves (2 M-groups x 4 N-groups); each wave: 32x32 x 4 gates (128 acc VGPRs).
// Weight slab (4 gates x 128 rows x 32 K = 32 KB) staged in LDS, double-buffered.
// ---------------------------------------------------------------------------
__global__ __launch_bounds__(256)
void lstm_enc_step(const bf16_t* __restrict__ Xb,    // [B, T, I] bf16
                   const bf16_t* __restrict__ Wenc,  // [4H, KENC] bf16
                   const float*  __restrict__ bias,  // [4H]
                   const bf16_t* __restrict__ h_in,  // [B, H]
                   bf16_t*       __restrict__ h_out, // [B, H]
                   float*        __restrict__ c,     // [B, H] fp32
                   int step) {
  __shared__ bf16_t sm[2][512 * 32];  // 2 x 32 KB

  const int lane = threadIdx.x & 31;
  const int wave = threadIdx.x >> 5;
  const int wm = wave & 1;
  const int wn = wave >> 1;
  const int mb = blockIdx.y * 64 + wm * 32;   // batch rows
  const int nb0 = blockIdx.x * 128;           // block's hidden-col base
  const int nb = nb0 + wn * 32;               // wave's hidden cols

  v8f zero = {};
  v8f acc[2][2][4];
#pragma unroll
  for (int mi = 0; mi < 2; ++mi)
#pragma unroll
    for (int ni = 0; ni < 2; ++ni)
#pragma unroll
      for (int g = 0; g < 4; ++g) acc[mi][ni][g] = zero;

  const bf16_t* Xs = Xb + (size_t)step * I_;
  const int NITER = KENC / 32;  // 40: 8 x-slabs then 32 h-slabs

  v8bf stage[8];
  fetch_slab<512>(stage, Wenc, KENC, nb0, 0);
  store_slab<512>(sm[0], stage);
  fetch_slab<512>(stage, Wenc, KENC, nb0, 32);

#pragma unroll 1
  for (int it = 0; it < NITER; ++it) {
    __syncthreads();  // sm[it&1] stores visible; prior readers of sm[(it+1)&1] done
    if (it + 1 < NITER) store_slab<512>(sm[(it + 1) & 1], stage);

    v16bf a0, a1;
    if (it < I_ / 32) {
      a0 = load_frag(Xs, mb,      T_ * I_, it * 32);
      a1 = load_frag(Xs, mb + 16, T_ * I_, it * 32);
    } else {
      a0 = load_frag(h_in, mb,      H_, it * 32 - I_);
      a1 = load_frag(h_in, mb + 16, H_, it * 32 - I_);
    }

    const bf16_t* sb = sm[it & 1];
#pragma unroll
    for (int g = 0; g < 4; ++g) {
      v16bf b0 = frag_from_lds(sb, g * 128 + wn * 32);
      v16bf b1 = frag_from_lds(sb, g * 128 + wn * 32 + 16);
      acc[0][0][g] = wmma_bf16(a0, b0, acc[0][0][g]);
      acc[0][1][g] = wmma_bf16(a0, b1, acc[0][1][g]);
      acc[1][0][g] = wmma_bf16(a1, b0, acc[1][0][g]);
      acc[1][1][g] = wmma_bf16(a1, b1, acc[1][1][g]);
    }

    if (it + 2 < NITER) fetch_slab<512>(stage, Wenc, KENC, nb0, (it + 2) * 32);
  }

  // ---- Fused gate epilogue: C/D layout lane->(n=L&15, m=r+8*(L>>4)) ------
#pragma unroll
  for (int ni = 0; ni < 2; ++ni) {
    const int n = nb + ni * 16 + (lane & 15);
    const float bi = bias[n];
    const float bf = bias[n + H_];
    const float bg = bias[n + 2 * H_];
    const float bo = bias[n + 3 * H_];
#pragma unroll
    for (int mi = 0; mi < 2; ++mi) {
#pragma unroll
      for (int r = 0; r < 8; ++r) {
        const int m = mb + mi * 16 + r + ((lane >> 4) << 3);
        const float ig = sigmoidf_(acc[mi][ni][0][r] + bi);
        const float fg = sigmoidf_(acc[mi][ni][1][r] + bf);
        const float gg = tanhf(acc[mi][ni][2][r] + bg);
        const float og = sigmoidf_(acc[mi][ni][3][r] + bo);
        const size_t idx = (size_t)m * H_ + n;
        const float cn = fg * c[idx] + ig * gg;
        c[idx] = cn;
        h_out[idx] = (bf16_t)(og * tanhf(cn));
      }
    }
  }
}

// ---------------------------------------------------------------------------
// One decoder LSTM step: z = [h, emb(y)] @ Wdec^T + b.  Wdec packed [4H,1056]
// with cols 0..1023 = Whh, 1024..1031 = Wih (K over E=8), 1032..1055 = 0.
// Last K-slab's A is the embedding gather (one b128 per lane, lanes 0-15).
// ---------------------------------------------------------------------------
__global__ __launch_bounds__(256)
void lstm_dec_step(const bf16_t* __restrict__ Emb,   // [V, E] bf16
                   const bf16_t* __restrict__ Wdec,  // [4H, KDEC] bf16
                   const float*  __restrict__ bias,  // [4H]
                   const bf16_t* __restrict__ h_in,  // [B, H]
                   bf16_t*       __restrict__ h_out, // [B, H]
                   float*        __restrict__ c,     // [B, H]
                   const int*    __restrict__ y) {   // [B] prev tokens
  __shared__ bf16_t sm[2][512 * 32];

  const int lane = threadIdx.x & 31;
  const int wave = threadIdx.x >> 5;
  const int wm = wave & 1;
  const int wn = wave >> 1;
  const int mb = blockIdx.y * 64 + wm * 32;
  const int nb0 = blockIdx.x * 128;
  const int nb = nb0 + wn * 32;

  v8f zero = {};
  v8f acc[2][2][4];
#pragma unroll
  for (int mi = 0; mi < 2; ++mi)
#pragma unroll
    for (int ni = 0; ni < 2; ++ni)
#pragma unroll
      for (int g = 0; g < 4; ++g) acc[mi][ni][g] = zero;

  const int NITER = KDEC / 32;  // 33: 32 h-slabs + 1 embedding slab

  v8bf stage[8];
  fetch_slab<512>(stage, Wdec, KDEC, nb0, 0);
  store_slab<512>(sm[0], stage);
  fetch_slab<512>(stage, Wdec, KDEC, nb0, 32);

#pragma unroll 1
  for (int it = 0; it < NITER; ++it) {
    __syncthreads();
    if (it + 1 < NITER) store_slab<512>(sm[(it + 1) & 1], stage);

    v16bf a0, a1;
    if (it < H_ / 32) {
      a0 = load_frag(h_in, mb,      H_, it * 32);
      a1 = load_frag(h_in, mb + 16, H_, it * 32);
    } else {
      // Embedding slab: lanes 0-15 hold k=1024..1031 (one emb row, 16B) in the
      // low half; everything else in the slab is zero (matches Wdec padding).
      v16bf z16 = {};
      a0 = z16; a1 = z16;
      if (lane < 16) {
        const int yv0 = y[mb + lane];
        const int yv1 = y[mb + 16 + lane];
        v8bf e0 = *(const v8bf*)(Emb + (size_t)yv0 * E_);
        v8bf e1 = *(const v8bf*)(Emb + (size_t)yv1 * E_);
        v8bf z8 = {};
        a0 = __builtin_shufflevector(e0, z8,
               0, 1, 2, 3, 4, 5, 6, 7, 8, 9, 10, 11, 12, 13, 14, 15);
        a1 = __builtin_shufflevector(e1, z8,
               0, 1, 2, 3, 4, 5, 6, 7, 8, 9, 10, 11, 12, 13, 14, 15);
      }
    }

    const bf16_t* sb = sm[it & 1];
#pragma unroll
    for (int g = 0; g < 4; ++g) {
      v16bf b0 = frag_from_lds(sb, g * 128 + wn * 32);
      v16bf b1 = frag_from_lds(sb, g * 128 + wn * 32 + 16);
      acc[0][0][g] = wmma_bf16(a0, b0, acc[0][0][g]);
      acc[0][1][g] = wmma_bf16(a0, b1, acc[0][1][g]);
      acc[1][0][g] = wmma_bf16(a1, b0, acc[1][0][g]);
      acc[1][1][g] = wmma_bf16(a1, b1, acc[1][1][g]);
    }

    if (it + 2 < NITER) fetch_slab<512>(stage, Wdec, KDEC, nb0, (it + 2) * 32);
  }

#pragma unroll
  for (int ni = 0; ni < 2; ++ni) {
    const int n = nb + ni * 16 + (lane & 15);
    const float bi = bias[n];
    const float bf = bias[n + H_];
    const float bg = bias[n + 2 * H_];
    const float bo = bias[n + 3 * H_];
#pragma unroll
    for (int mi = 0; mi < 2; ++mi) {
#pragma unroll
      for (int r = 0; r < 8; ++r) {
        const int m = mb + mi * 16 + r + ((lane >> 4) << 3);
        const float ig = sigmoidf_(acc[mi][ni][0][r] + bi);
        const float fg = sigmoidf_(acc[mi][ni][1][r] + bf);
        const float gg = tanhf(acc[mi][ni][2][r] + bg);
        const float og = sigmoidf_(acc[mi][ni][3][r] + bo);
        const size_t idx = (size_t)m * H_ + n;
        const float cn = fg * c[idx] + ig * gg;
        c[idx] = cn;
        h_out[idx] = (bf16_t)(og * tanhf(cn));
      }
    }
  }
}

// ---------------------------------------------------------------------------
// logits[B, V] = h @ fcW^T + fc_b, written into d_out[:, step, :] (fp32).
// Same staged-B scheme with a 128-row slab (2 x 8 KB LDS).
// ---------------------------------------------------------------------------
__global__ __launch_bounds__(256)
void fc_logits(const bf16_t* __restrict__ h,    // [B, H]
               const bf16_t* __restrict__ Wfc,  // [V, H]
               const float*  __restrict__ bias, // [V]
               float*        __restrict__ out,  // [B, FUT, V]
               int step) {
  __shared__ bf16_t sm[2][128 * 32];

  const int lane = threadIdx.x & 31;
  const int wave = threadIdx.x >> 5;
  const int wm = wave & 1;
  const int wn = wave >> 1;
  const int mb = blockIdx.y * 64 + wm * 32;
  const int nb0 = blockIdx.x * 128;
  const int nb = nb0 + wn * 32;

  v8f zero = {};
  v8f acc[2][2];
  acc[0][0] = zero; acc[0][1] = zero; acc[1][0] = zero; acc[1][1] = zero;

  const int NITER = H_ / 32;  // 32

  v8bf stage[2];
  fetch_slab<128>(stage, Wfc, H_, nb0, 0);
  store_slab<128>(sm[0], stage);
  fetch_slab<128>(stage, Wfc, H_, nb0, 32);

#pragma unroll 1
  for (int it = 0; it < NITER; ++it) {
    __syncthreads();
    if (it + 1 < NITER) store_slab<128>(sm[(it + 1) & 1], stage);

    v16bf a0 = load_frag(h, mb,      H_, it * 32);
    v16bf a1 = load_frag(h, mb + 16, H_, it * 32);

    const bf16_t* sb = sm[it & 1];
    v16bf b0 = frag_from_lds(sb, wn * 32);
    v16bf b1 = frag_from_lds(sb, wn * 32 + 16);
    acc[0][0] = wmma_bf16(a0, b0, acc[0][0]);
    acc[0][1] = wmma_bf16(a0, b1, acc[0][1]);
    acc[1][0] = wmma_bf16(a1, b0, acc[1][0]);
    acc[1][1] = wmma_bf16(a1, b1, acc[1][1]);

    if (it + 2 < NITER) fetch_slab<128>(stage, Wfc, H_, nb0, (it + 2) * 32);
  }

#pragma unroll
  for (int ni = 0; ni < 2; ++ni) {
    const int n = nb + ni * 16 + (lane & 15);
    const float bn = bias[n];
#pragma unroll
    for (int mi = 0; mi < 2; ++mi) {
#pragma unroll
      for (int r = 0; r < 8; ++r) {
        const int m = mb + mi * 16 + r + ((lane >> 4) << 3);
        out[((size_t)m * FUT_ + step) * V_ + n] = acc[mi][ni][r] + bn;
      }
    }
  }
}

// ---------------------------------------------------------------------------
// Row argmax over V=1024 logits (first occurrence on ties, like jnp.argmax).
// ---------------------------------------------------------------------------
__global__ __launch_bounds__(256)
void argmax_rows(const float* __restrict__ out, int step,
                 int* __restrict__ y) {
  __shared__ float sv[256];
  __shared__ int   si[256];
  const int b = blockIdx.x;
  const int t = threadIdx.x;
  const float* row = out + ((size_t)b * FUT_ + step) * V_;
  float best = -3.402823466e38f;
  int bidx = V_;
  for (int k = t; k < V_; k += 256) {
    const float v = row[k];
    if (v > best || (v == best && k < bidx)) { best = v; bidx = k; }
  }
  sv[t] = best; si[t] = bidx;
  __syncthreads();
  for (int off = 128; off > 0; off >>= 1) {
    if (t < off) {
      const float v2 = sv[t + off];
      const int   i2 = si[t + off];
      if (v2 > sv[t] || (v2 == sv[t] && i2 < si[t])) { sv[t] = v2; si[t] = i2; }
    }
    __syncthreads();
  }
  if (t == 0) y[b] = si[0];
}

// ---------------------------------------------------------------------------
// One-time (per launch) weight / input conversion + state init kernels.
// ---------------------------------------------------------------------------
__global__ void cvt_f32_to_bf16(const float* __restrict__ src,
                                bf16_t* __restrict__ dst, int n) {
  const int i = blockIdx.x * blockDim.x + threadIdx.x;
  if (i < n) dst[i] = (bf16_t)src[i];
}

__global__ void pack_encW(const float* __restrict__ Wih,   // [4H, I]
                          const float* __restrict__ Whh,   // [4H, H]
                          bf16_t* __restrict__ dst) {      // [4H, KENC]
  const int i = blockIdx.x * blockDim.x + threadIdx.x;
  if (i >= 4 * H_ * KENC) return;
  const int row = i / KENC, col = i % KENC;
  const float v = (col < I_) ? Wih[row * I_ + col]
                             : Whh[row * H_ + (col - I_)];
  dst[i] = (bf16_t)v;
}

__global__ void pack_decW(const float* __restrict__ Wih,   // [4H, E]
                          const float* __restrict__ Whh,   // [4H, H]
                          bf16_t* __restrict__ dst) {      // [4H, KDEC]
  const int i = blockIdx.x * blockDim.x + threadIdx.x;
  if (i >= 4 * H_ * KDEC) return;
  const int row = i / KDEC, col = i % KDEC;
  float v = 0.0f;
  if (col < H_)           v = Whh[row * H_ + col];
  else if (col < H_ + E_) v = Wih[row * E_ + (col - H_)];
  dst[i] = (bf16_t)v;
}

__global__ void init_state(float* __restrict__ c, bf16_t* __restrict__ h0,
                           int* __restrict__ y) {
  const int i = blockIdx.x * blockDim.x + threadIdx.x;
  if (i < B_ * H_) { c[i] = 0.0f; h0[i] = (bf16_t)0.0f; }
  if (i < B_) y[i] = 0;
}

// ---------------------------------------------------------------------------
// Host launcher.
// ---------------------------------------------------------------------------
extern "C" void kernel_launch(void* const* d_in, const int* in_sizes, int n_in,
                              void* d_out, int out_size, void* d_ws,
                              size_t ws_size, hipStream_t stream) {
  (void)in_sizes; (void)n_in; (void)out_size; (void)ws_size;

  const float* x_hist  = (const float*)d_in[0];  // [B, T, I]
  const float* enc_Wih = (const float*)d_in[1];  // [4H, I]
  const float* enc_Whh = (const float*)d_in[2];  // [4H, H]
  const float* enc_b   = (const float*)d_in[3];  // [4H]
  const float* embed_W = (const float*)d_in[4];  // [V, E]
  const float* dec_Wih = (const float*)d_in[5];  // [4H, E]
  const float* dec_Whh = (const float*)d_in[6];  // [4H, H]
  const float* dec_b   = (const float*)d_in[7];  // [4H]
  const float* fc_W    = (const float*)d_in[8];  // [V, H]
  const float* fc_b    = (const float*)d_in[9];  // [V]
  float* out = (float*)d_out;                    // [B, FUT, V]

  char* ws = (char*)d_ws;
  size_t off = 0;
  auto carve = [&](size_t bytes) {
    char* p = ws + off;
    off += (bytes + 255) & ~(size_t)255;
    return p;
  };
  bf16_t* Wenc = (bf16_t*)carve((size_t)4 * H_ * KENC * 2);  // 10.5 MB
  bf16_t* Wdec = (bf16_t*)carve((size_t)4 * H_ * KDEC * 2);  //  8.7 MB
  bf16_t* Wfc  = (bf16_t*)carve((size_t)V_ * H_ * 2);        //  2.0 MB
  bf16_t* Emb  = (bf16_t*)carve((size_t)V_ * E_ * 2);        //   16 KB
  bf16_t* Xb   = (bf16_t*)carve((size_t)B_ * T_ * I_ * 2);   // 67.1 MB
  bf16_t* h0   = (bf16_t*)carve((size_t)B_ * H_ * 2);        //  1.0 MB
  bf16_t* h1   = (bf16_t*)carve((size_t)B_ * H_ * 2);        //  1.0 MB
  float*  cst  = (float*) carve((size_t)B_ * H_ * 4);        //  2.0 MB
  int*    yb   = (int*)   carve((size_t)B_ * 4);             //  2.0 KB
  bf16_t* hbuf[2] = {h0, h1};

  const dim3 blk(256);

  // --- conversions / init (run every call; deterministic, cheap) ----------
  {
    int n;
    n = 4 * H_ * KENC;
    pack_encW<<<dim3((n + 255) / 256), blk, 0, stream>>>(enc_Wih, enc_Whh, Wenc);
    n = 4 * H_ * KDEC;
    pack_decW<<<dim3((n + 255) / 256), blk, 0, stream>>>(dec_Wih, dec_Whh, Wdec);
    n = V_ * H_;
    cvt_f32_to_bf16<<<dim3((n + 255) / 256), blk, 0, stream>>>(fc_W, Wfc, n);
    n = V_ * E_;
    cvt_f32_to_bf16<<<dim3((n + 255) / 256), blk, 0, stream>>>(embed_W, Emb, n);
    n = B_ * T_ * I_;
    cvt_f32_to_bf16<<<dim3((n + 255) / 256), blk, 0, stream>>>(x_hist, Xb, n);
    n = B_ * H_;
    init_state<<<dim3((n + 255) / 256), blk, 0, stream>>>(cst, h0, yb);
  }

  // --- encoder: 256 sequential steps, ping-pong h -------------------------
  const dim3 gcell(8, 8);  // x: H/128 col-blocks, y: B/64 row-blocks
  for (int s = 0; s < T_; ++s) {
    lstm_enc_step<<<gcell, blk, 0, stream>>>(
        Xb, Wenc, enc_b, hbuf[s & 1], hbuf[(s + 1) & 1], cst, s);
  }

  // --- decoder: 64 sequential steps (cell -> logits -> argmax feedback) ---
  for (int d = 0; d < FUT_; ++d) {
    lstm_dec_step<<<gcell, blk, 0, stream>>>(
        Emb, Wdec, dec_b, hbuf[d & 1], hbuf[(d + 1) & 1], cst, yb);
    fc_logits<<<gcell, blk, 0, stream>>>(
        hbuf[(d + 1) & 1], Wfc, fc_b, out, d);
    argmax_rows<<<dim3(B_), blk, 0, stream>>>(out, d, yb);
  }
}
